// _TinyRNN_2594160247513
// MI455X (gfx1250) — compile-verified, matrix operand order checked by reading, not run on hardware
//
#include <hip/hip_runtime.h>
#include <math.h>

// ---------------- problem constants ----------------
constexpr int Bsz = 2048;
constexpr int Tt  = 256;
constexpr int Ff  = 64;   // input features
constexpr int Hh  = 32;   // hidden
// 3H = 96 gate cols = 6 tiles of 16 (tiles 0,1 = r; 2,3 = z; 4,5 = n)
// Block: 4 waves. Waves 0,1 = consumers (h recurrence, gate-half 0/1).
//                 Waves 2,3 = producers (x-side WMMAs, gate-half 0/1), 1 step ahead.

typedef __attribute__((ext_vector_type(16))) __bf16 v16bf;
typedef __attribute__((ext_vector_type(8)))  float  v8f;
typedef __attribute__((ext_vector_type(4)))  float  f4;
typedef int v4i __attribute__((vector_size(16)));   // matches builtin param type

#define AS_GLOBAL __attribute__((address_space(1)))
#define AS_LDS    __attribute__((address_space(3)))

#define XSTRIDE 68   // 64 floats + 4 pad
#define HSTRIDE 36   // 32 floats + 4 pad
#define XGSTR   12   // xg fragment: 8 floats/lane + 4 pad (16B-aligned slots)

// ---- CDNA5 async global->LDS path (ASYNCcnt), with fallback ----
#if __has_builtin(__builtin_amdgcn_global_load_async_to_lds_b128)
#define HAVE_ASYNC_LDS 1
#else
#define HAVE_ASYNC_LDS 0
#endif

#if HAVE_ASYNC_LDS
# if __has_builtin(__builtin_amdgcn_s_wait_asynccnt)
#  define WAIT_ASY(N) __builtin_amdgcn_s_wait_asynccnt(N)
# else
#  define WAIT_ASY(N) asm volatile("s_wait_asynccnt %0" :: "i"(N) : "memory")
# endif
#else
# define WAIT_ASY(N) asm volatile("s_wait_loadcnt 0\n\ts_wait_dscnt 0" ::: "memory")
#endif
#define CFENCE() asm volatile("" ::: "memory")

__device__ __forceinline__ v8f splat8(float s) {
  return (v8f){s, s, s, s, s, s, s, s};
}
__device__ __forceinline__ float rcp_fast(float x) {
  return __builtin_amdgcn_rcpf(x);
}
__device__ __forceinline__ float sigmoid_fast(float x) {
  return rcp_fast(1.0f + __expf(-x));
}
__device__ __forceinline__ float tanh_fast(float x) {
  const float e = __expf(-2.0f * x);
  return (1.0f - e) * rcp_fast(1.0f + e);
}

// Producer stages its own FULL 16x64 f32 tile (8 async B128 per lane):
// lane>>1 = row, (lane&1)*32 = column half.
__device__ __forceinline__ void stage_x_tile(const float* __restrict__ gsrc,
                                             float* ldst) {
#if HAVE_ASYNC_LDS
#pragma unroll
  for (int i = 0; i < 8; ++i) {
    __builtin_amdgcn_global_load_async_to_lds_b128(
        (AS_GLOBAL v4i*)(gsrc + i * 4),
        (AS_LDS v4i*)(ldst + i * 4), 0, 0);
  }
#else
#pragma unroll
  for (int i = 0; i < 8; ++i) {
    f4 v = *(const f4*)(gsrc + i * 4);
    *(f4*)(ldst + i * 4) = v;
  }
#endif
}

#define WMMA_BF16(A, B, C) \
  __builtin_amdgcn_wmma_f32_16x16x32_bf16(false, (A), false, (B), (short)0, (C), false, false)

__global__ __launch_bounds__(128) void gru_fused(
    const float* __restrict__ x,       // [B,T,F]
    const float* __restrict__ W_ih,    // [96,64]
    const float* __restrict__ W_hh,    // [96,32]
    const float* __restrict__ b_ih,    // [96]
    const float* __restrict__ b_hh,    // [96]
    const float* __restrict__ W_head,  // [1,32]
    const float* __restrict__ b_head,  // [1]
    float* __restrict__ out)           // [B,1]
{
  __shared__ float xbuf[2][2][16 * XSTRIDE];        // [producer][buf]
  __shared__ float hbuf[2][16 * HSTRIDE];           // [buf]
  __shared__ float xgbuf[2][2][3][32 * XGSTR];      // [buf][half][jj][lane slot]

  const int tid  = threadIdx.x;
  const int wid  = tid >> 5;
  const int lane = tid & 31;
  const int n    = lane & 15;
  const int kg   = lane >> 4;
  const int b0   = blockIdx.x * 16;
  const int half = wid & 1;              // gate-column half for this wave
  const bool is_prod = (wid >= 2);

  if (is_prod) {
    // =================== PRODUCER: x-side WMMAs, 1 step ahead ===================
    v16bf Wih[2][3];   // [K-tile][jj], tile j = 2*jj + half
#pragma unroll
    for (int jj = 0; jj < 3; ++jj) {
      const int j = 2 * jj + half;
#pragma unroll
      for (int e = 0; e < 16; ++e) {
        const int kl = (e < 8) ? (kg * 8 + e) : (16 + kg * 8 + (e - 8));
        Wih[0][jj][e] = (__bf16)W_ih[(16 * j + n) * Ff + kl];
        Wih[1][jj][e] = (__bf16)W_ih[(16 * j + n) * Ff + 32 + kl];
      }
    }
    // r,z accumulators carry BOTH x- and h-bias; n carries only x-bias
    const float bR  = b_ih[     16 * half + n] + b_hh[     16 * half + n];
    const float bZ  = b_ih[32 + 16 * half + n] + b_hh[32 + 16 * half + n];
    const float bxN = b_ih[64 + 16 * half + n];

    const int xr = lane >> 1;
    const int xh = (lane & 1) * 32;
    const float* gx = x + ((size_t)(b0 + xr) * Tt) * Ff + xh;
    float* lx0 = &xbuf[half][0][xr * XSTRIDE + xh];
    float* lx1 = &xbuf[half][1][xr * XSTRIDE + xh];

    // compute xg(step) from xbuf[half][srcbuf] into xgbuf[srcbuf... caller picks]
    auto do_xg = [&](const float* xs, float* dst0, float* dst1, float* dst2) {
      v16bf xa0, xa1;
#pragma unroll
      for (int e = 0; e < 16; ++e) {
        const int kl = (e < 8) ? (kg * 8 + e) : (16 + kg * 8 + (e - 8));
        xa0[e] = (__bf16)xs[n * XSTRIDE + kl];
        xa1[e] = (__bf16)xs[n * XSTRIDE + 32 + kl];
      }
      v8f aR = splat8(bR), aZ = splat8(bZ), xN = splat8(bxN);
      aR = WMMA_BF16(xa0, Wih[0][0], aR);  aR = WMMA_BF16(xa1, Wih[1][0], aR);
      aZ = WMMA_BF16(xa0, Wih[0][1], aZ);  aZ = WMMA_BF16(xa1, Wih[1][1], aZ);
      xN = WMMA_BF16(xa0, Wih[0][2], xN);  xN = WMMA_BF16(xa1, Wih[1][2], xN);
      *(f4*)(dst0)     = (f4){aR[0], aR[1], aR[2], aR[3]};
      *(f4*)(dst0 + 4) = (f4){aR[4], aR[5], aR[6], aR[7]};
      *(f4*)(dst1)     = (f4){aZ[0], aZ[1], aZ[2], aZ[3]};
      *(f4*)(dst1 + 4) = (f4){aZ[4], aZ[5], aZ[6], aZ[7]};
      *(f4*)(dst2)     = (f4){xN[0], xN[1], xN[2], xN[3]};
      *(f4*)(dst2 + 4) = (f4){xN[4], xN[5], xN[6], xN[7]};
    };

    // prologue: xg(0) ready before first barrier; x(1) staging in flight
    stage_x_tile(gx, lx0);
    WAIT_ASY(0);
    CFENCE();
    do_xg(xbuf[half][0],
          &xgbuf[0][half][0][lane * XGSTR],
          &xgbuf[0][half][1][lane * XGSTR],
          &xgbuf[0][half][2][lane * XGSTR]);
    stage_x_tile(gx + (size_t)Ff, lx1);
    __syncthreads();

    for (int t = 0; t < Tt; ++t) {
      const int cur = t & 1, nxt = cur ^ 1;
      if (t + 2 < Tt) stage_x_tile(gx + (size_t)(t + 2) * Ff, cur ? lx1 : lx0);
      if (t + 1 < Tt) {
        if (t + 2 < Tt) WAIT_ASY(8); else WAIT_ASY(0);  // x(t+1) landed
        CFENCE();
        do_xg(xbuf[half][nxt],
              &xgbuf[nxt][half][0][lane * XGSTR],
              &xgbuf[nxt][half][1][lane * XGSTR],
              &xgbuf[nxt][half][2][lane * XGSTR]);
      }
      __syncthreads();
    }
  } else {
    // =================== CONSUMER: h recurrence (serial path) ===================
    v16bf Whh[3];
#pragma unroll
    for (int jj = 0; jj < 3; ++jj) {
      const int j = 2 * jj + half;
#pragma unroll
      for (int e = 0; e < 16; ++e) {
        const int kl = (e < 8) ? (kg * 8 + e) : (16 + kg * 8 + (e - 8));
        Whh[jj][e] = (__bf16)W_hh[(16 * j + n) * Hh + kl];
      }
    }
    const float bhN = b_hh[64 + 16 * half + n];

    v8f hc = splat8(0.0f);
    // h0 = 0 published for step 0
#pragma unroll
    for (int v = 0; v < 8; ++v)
      hbuf[0][(v + 8 * kg) * HSTRIDE + 16 * half + n] = 0.0f;
    __syncthreads();

    for (int t = 0; t < Tt; ++t) {
      const int cur = t & 1, nxt = cur ^ 1;
      // xg(t) fragments (pre-biased) from producer
      const float* sR = &xgbuf[cur][half][0][lane * XGSTR];
      const float* sZ = &xgbuf[cur][half][1][lane * XGSTR];
      const float* sN = &xgbuf[cur][half][2][lane * XGSTR];
      const f4 r0 = *(const f4*)(sR), r1 = *(const f4*)(sR + 4);
      const f4 z0 = *(const f4*)(sZ), z1 = *(const f4*)(sZ + 4);
      const f4 n0 = *(const f4*)(sN), n1 = *(const f4*)(sN + 4);
      v8f aR = (v8f){r0.x, r0.y, r0.z, r0.w, r1.x, r1.y, r1.z, r1.w};
      v8f aZ = (v8f){z0.x, z0.y, z0.z, z0.w, z1.x, z1.y, z1.z, z1.w};
      v8f xN = (v8f){n0.x, n0.y, n0.z, n0.w, n1.x, n1.y, n1.z, n1.w};

      // full h_t row as bf16 A-fragment
      v16bf ha;
#pragma unroll
      for (int e = 0; e < 16; ++e) {
        const int kl = (e < 8) ? (kg * 8 + e) : (16 + kg * 8 + (e - 8));
        ha[e] = (__bf16)hbuf[cur][n * HSTRIDE + kl];
      }

      // the 3 serial-path WMMAs
      aR = WMMA_BF16(ha, Whh[0], aR);
      aZ = WMMA_BF16(ha, Whh[1], aZ);
      v8f hN = splat8(bhN);
      hN = WMMA_BF16(ha, Whh[2], hN);

      // gates: h = n + z*(h - n), n = tanh(xn + r*hn)
#pragma unroll
      for (int v = 0; v < 8; ++v) {
        const float r  = sigmoid_fast(aR[v]);
        const float z  = sigmoid_fast(aZ[v]);
        const float nn = tanh_fast(xN[v] + r * hN[v]);
        hc[v] = nn + z * (hc[v] - nn);
      }

      // publish h_{t+1}
#pragma unroll
      for (int v = 0; v < 8; ++v)
        hbuf[nxt][(v + 8 * kg) * HSTRIDE + 16 * half + n] = hc[v];
      __syncthreads();
    }
  }

  // ---------------- head: y = sigmoid(h_T . W_head^T + b_head) ----------------
  // final h_T is in hbuf[0] (iter 255 stored to buf (255+1)&1 == 0), barrier done
  if (tid < 16) {
    float acc = b_head[0];
#pragma unroll
    for (int c = 0; c < Hh; ++c) acc += hbuf[0][tid * HSTRIDE + c] * W_head[c];
    out[b0 + tid] = sigmoid_fast(acc);
  }
}

extern "C" void kernel_launch(void* const* d_in, const int* in_sizes, int n_in,
                              void* d_out, int out_size, void* d_ws, size_t ws_size,
                              hipStream_t stream) {
  (void)in_sizes; (void)n_in; (void)d_ws; (void)ws_size; (void)out_size;
  const float* x      = (const float*)d_in[0];
  const float* W_ih   = (const float*)d_in[1];
  const float* W_hh   = (const float*)d_in[2];
  const float* b_ih   = (const float*)d_in[3];
  const float* b_hh   = (const float*)d_in[4];
  const float* W_head = (const float*)d_in[5];
  const float* b_head = (const float*)d_in[6];
  float* out = (float*)d_out;

  dim3 grid(Bsz / 16);  // 128 workgroups x 4 waves (one 16-row tile each)
  dim3 block(128);
  gru_fused<<<grid, block, 0, stream>>>(x, W_ih, W_hh, b_ih, b_hh,
                                        W_head, b_head, out);
}
